// PositionAttention_43954695307427
// MI455X (gfx1250) — compile-verified, matrix operand order checked by reading, not run on hardware
//
#include <hip/hip_runtime.h>
#include <hip/hip_bf16.h>

// ---------------------------------------------------------------------------
// PositionAttention fused kernels for MI455X (gfx1250, wave32, WMMA)
//
// Shapes: BS=32, C=64, H*W=4096, NP=512.
//   k[b]   = Ua_w(512x64) @ key[b](64x4096)        (1x1 conv as GEMM)
//   score  = tanh(k + q[b,:,None])
//   e_t    = Va_w(512) . score + Va_b              ([b, hw])
//   alpha  = exp(e_t) / (sum_hw exp(e_t) + eps)    (no max-subtraction)
//   out    = alpha[:,None,:] * feature             ([b, 64, 4096])
//
// Memory floor ~100 MB => ~4.3us @ 23.3 TB/s. GEMM (8.6 GFLOP) runs on
// v_wmma_f32_16x16x32_f16; tanh runs on the CDNA5 V_TANH_F32 TRANS unit so
// the epilogue co-executes with the matrix pipe. Each wave carries TWO M=16
// tiles so every B-fragment load feeds two WMMAs (B-load traffic halved).
// ---------------------------------------------------------------------------

typedef _Float16 v16h __attribute__((ext_vector_type(16)));
typedef _Float16 h8   __attribute__((ext_vector_type(8)));
typedef float    v8f  __attribute__((ext_vector_type(8)));

#define BS_  32
#define C_   64
#define HW_  4096
#define NP_  512
#define EPS_ 1e-8f

// Workspace layout (float offsets):
//   e_exp : [0, BS*HW)                      = 131072 floats
//   qb    : [131072, 131072+BS*NP)          = 16384 floats (query + Ua_b)
//   sums  : [147456, 147456+BS)             = 32 floats
//   Ua_h  : float offset 147488 as _Float16*, NP*64 halves (16B aligned)
#define WS_QB_   131072
#define WS_SUMS_ 147456
#define WS_UAH_  147488

// --- Fast branch-free tanh: native V_TANH_F32 when the builtin exists. ---
__device__ __forceinline__ float fast_tanh(float x) {
#if __has_builtin(__builtin_amdgcn_tanhf)
    return __builtin_amdgcn_tanhf(x);
#elif __has_builtin(__builtin_amdgcn_tanh_f32)
    return __builtin_amdgcn_tanh_f32(x);
#else
    // Branch-free fallback: tanh(x) = sign(x) * (1 - e) / (1 + e), e = exp(-2|x|).
    const float ax = __builtin_fabsf(x);
    const float e  = __expf(-2.0f * ax);
    const float r  = (1.0f - e) * __frcp_rn(1.0f + e);
    return __builtin_copysignf(r, x);
#endif
}

// ---------------------------------------------------------------------------
// Prep kernel (tiny): f16 copy of Ua_w, fold Ua_b into query, zero sums.
// Launched as 128 blocks x 256 threads = 32768 = NP*64 lanes.
// ---------------------------------------------------------------------------
__global__ __launch_bounds__(256) void pa_prep(
    const float* __restrict__ Ua_w,   // [NP, 64]
    const float* __restrict__ Ua_b,   // [NP]
    const float* __restrict__ query,  // [BS, 1, NP]
    _Float16* __restrict__ Ua_h,      // [NP, 64] f16
    float* __restrict__ qb,           // [BS, NP]
    float* __restrict__ sums)         // [BS]
{
    const int i = blockIdx.x * 256 + threadIdx.x;     // 0 .. NP*64-1
    Ua_h[i] = (_Float16)Ua_w[i];
    if (i < BS_ * NP_) qb[i] = query[i] + Ua_b[i & (NP_ - 1)];
    if (i < BS_)       sums[i] = 0.0f;
}

// ---------------------------------------------------------------------------
// Kernel 1: e_exp[b,hw] = exp(e_t[b,hw]); partial sums atomically folded into
// sums[b]. One wave owns 2 x (M=16) WMMA tiles = 32 spatial positions, which
// share every B fragment. Block = 128 threads = 4 waves = 128 positions;
// grid = (HW/128, BS).
//
// Fragment layouts (CDNA5 ISA 7.12.2, 16-bit operands, wave32):
//   A (16x32, MxK): lane%16 = M row; half h: K = (h&7) + 16*(h>>3) + 8*(lane>>4)
//   B (32x16, KxN): lane%16 = N col; same K striping per half/lane-group
//   C/D (16x16 f32): lane%16 = N col; VGPR i = M row (lanes>=16: M = i+8)
// ---------------------------------------------------------------------------
__global__ __launch_bounds__(128) void pa_score_kernel(
    const float*    __restrict__ key,    // [BS, 64, HW]
    const _Float16* __restrict__ Ua_h,   // [NP, 64] f16 (prepped)
    const float*    __restrict__ qb,     // [BS, NP]     (query + Ua_b)
    const float*    __restrict__ Va_w,   // [1, NP]
    const float*    __restrict__ Va_b,   // [1]
    float* __restrict__ e_exp,           // [BS, HW]
    float* __restrict__ sums)            // [BS]
{
    const int b      = blockIdx.y;
    const int wave   = threadIdx.x >> 5;
    const int lane   = threadIdx.x & 31;
    const int hwbase = blockIdx.x * 128 + wave * 32;   // 32 positions per wave
    const int m      = lane & 15;   // M row (A) / N col (B, C/D)
    const int grp    = lane >> 4;   // K (+8) for A/B halves; M (+8) for C/D

    // --- A fragments: two key tiles [M=16 x K=64] each, loaded once.
    // key is channel-major; lanes 0..15 read consecutive hw -> coalesced.
    v16h a0[2], a1[2];
#pragma unroll
    for (int t = 0; t < 2; ++t) {
        const float* kptr = key + (size_t)b * C_ * HW_
                                + (size_t)(hwbase + t * 16 + m);
#pragma unroll
        for (int h = 0; h < 16; ++h) {
            const int k = (h & 7) + 16 * (h >> 3) + 8 * grp;   // 0..31
            a0[t][h] = (_Float16)kptr[(size_t)k * HW_];
            a1[t][h] = (_Float16)kptr[(size_t)(k + 32) * HW_];
        }
    }

    float acc[2][8];
#pragma unroll
    for (int t = 0; t < 2; ++t)
#pragma unroll
        for (int i = 0; i < 8; ++i) acc[t][i] = 0.0f;

    // --- Loop over the 512 conv outputs in N=16 tiles (32 iterations).
    // B fragments are native f16 (16B-aligned b128 loads) and feed BOTH
    // M-tiles: 8 WMMA per ~12 VMEM ops in the steady state.
    for (int nb = 0; nb < NP_; nb += 16) {
        const int n = nb + m;                        // this lane's output chan
        const h8* up = (const h8*)(Ua_h + (size_t)n * 64 + 8 * grp);
        const h8 q0 = up[0];   // k = 8g      .. +7
        const h8 q1 = up[2];   // k = 8g + 16 .. +7
        const h8 q2 = up[4];   // k = 8g + 32 .. +7
        const h8 q3 = up[6];   // k = 8g + 48 .. +7
        const v16h bf0 = __builtin_shufflevector(q0, q1, 0,1,2,3,4,5,6,7,
                                                 8,9,10,11,12,13,14,15);
        const v16h bf1 = __builtin_shufflevector(q2, q3, 0,1,2,3,4,5,6,7,
                                                 8,9,10,11,12,13,14,15);

        const float qn  = qb[b * NP_ + n];
        const float van = Va_w[n];

#pragma unroll
        for (int t = 0; t < 2; ++t) {
            v8f c = {};
            c = __builtin_amdgcn_wmma_f32_16x16x32_f16(false, a0[t], false, bf0,
                                                       (short)0, c, false, false);
            c = __builtin_amdgcn_wmma_f32_16x16x32_f16(false, a1[t], false, bf1,
                                                       (short)0, c, false, false);
            // Branch-free epilogue: column n fixed per lane; TANH on TRANS pipe.
#pragma unroll
            for (int i = 0; i < 8; ++i)
                acc[t][i] = __builtin_fmaf(van, fast_tanh(c[i] + qn), acc[t][i]);
        }
    }

    // --- Reduce over the 16 N-columns held by each 16-lane half.
    // xor masks 1..8 never cross the 16-lane half boundary, so lanes 0-15
    // (M=0..7) and lanes 16-31 (M=8..15) reduce independently.
#pragma unroll
    for (int mask = 1; mask <= 8; mask <<= 1)
#pragma unroll
        for (int t = 0; t < 2; ++t)
#pragma unroll
            for (int i = 0; i < 8; ++i)
                acc[t][i] += __shfl_xor(acc[t][i], mask, 32);

    // --- Writer lane per half: exp + store + fold into batch denominator.
    if (m == 0) {
        const float vb = Va_b[0];
        float lsum = 0.0f;
#pragma unroll
        for (int t = 0; t < 2; ++t)
#pragma unroll
            for (int i = 0; i < 8; ++i) {
                const float e = __expf(acc[t][i] + vb);   // no max-sub (matches ref)
                e_exp[(size_t)b * HW_ + hwbase + t * 16 + grp * 8 + i] = e;
                lsum += e;
            }
        atomicAdd(&sums[b], lsum);
    }
}

// ---------------------------------------------------------------------------
// Kernel 2: out[b,c,hw] = feature[b,c,hw] * e_exp[b,hw] / (sums[b] + eps).
// Pure streaming; float4 (global_load_b128 / global_store_b128).
// ---------------------------------------------------------------------------
__global__ __launch_bounds__(256) void pa_weight_kernel(
    const float* __restrict__ feature,   // [BS, C, HW]
    const float* __restrict__ e_exp,     // [BS, HW]
    const float* __restrict__ sums,      // [BS]
    float* __restrict__ out)             // [BS, C, HW]
{
    const size_t i4  = (size_t)blockIdx.x * blockDim.x + threadIdx.x;
    const size_t idx = i4 * 4;                       // flat float index, %4 == 0
    const int    b   = (int)(idx / (C_ * HW_));
    const int    hw  = (int)(idx % HW_);             // 4 consecutive hw

    const float inv = 1.0f / (sums[b] + EPS_);
    const float4 f  = *(const float4*)(feature + idx);
    const float4 e  = *(const float4*)(e_exp + (size_t)b * HW_ + hw);

    float4 o;
    o.x = f.x * (e.x * inv);
    o.y = f.y * (e.y * inv);
    o.z = f.z * (e.z * inv);
    o.w = f.w * (e.w * inv);
    *(float4*)(out + idx) = o;
}

// ---------------------------------------------------------------------------
// Host-side launcher. Inputs (setup_inputs order, all f32):
//   0: feature [32,64,32,128]  1: query [32,1,512]  2: key [32,64,32,128]
//   3: Ua_w [512,64]  4: Ua_b [512]  5: Va_w [1,512]  6: Va_b [1]
// ---------------------------------------------------------------------------
extern "C" void kernel_launch(void* const* d_in, const int* in_sizes, int n_in,
                              void* d_out, int out_size, void* d_ws, size_t ws_size,
                              hipStream_t stream) {
    const float* feature = (const float*)d_in[0];
    const float* query   = (const float*)d_in[1];
    const float* key     = (const float*)d_in[2];
    const float* Ua_w    = (const float*)d_in[3];
    const float* Ua_b    = (const float*)d_in[4];
    const float* Va_w    = (const float*)d_in[5];
    const float* Va_b    = (const float*)d_in[6];
    float*       out     = (float*)d_out;

    float*    wsf   = (float*)d_ws;
    float*    e_exp = wsf;                        // BS*HW floats
    float*    qb    = wsf + WS_QB_;               // BS*NP floats
    float*    sums  = wsf + WS_SUMS_;             // BS floats
    _Float16* Ua_h  = (_Float16*)(wsf + WS_UAH_); // NP*64 halves, 16B aligned

    // 1) prep: f16 weights, query+bias fold, zero sums (graph-safe every call).
    pa_prep<<<(NP_ * C_) / 256, 256, 0, stream>>>(Ua_w, Ua_b, query,
                                                  Ua_h, qb, sums);

    // 2) WMMA score kernel: grid (HW/128, BS) blocks of 128 threads.
    dim3 gridS(HW_ / 128, BS_);
    pa_score_kernel<<<gridS, 128, 0, stream>>>(key, Ua_h, qb,
                                               Va_w, Va_b, e_exp, sums);

    // 3) streaming weight kernel: BS*C*HW/4 float4 lanes.
    const int n4 = (BS_ * C_ * HW_) / 4;          // 2,097,152
    pa_weight_kernel<<<n4 / 256, 256, 0, stream>>>(feature, e_exp, sums, out);
}
